// HMSTA_v1b_GAT_3667902071298
// MI455X (gfx1250) — compile-verified, matrix operand order checked by reading, not run on hardware
//
#include <hip/hip_runtime.h>
#include <hip/hip_bf16.h>
#include <stdint.h>

#define N_NODES 100000
#define F_IN    166
#define KP_IN   192
#define HID     128
#define HEADS   4
#define NEG_SLOPE 0.2f

typedef __attribute__((ext_vector_type(16))) __bf16 v16bf;
typedef __attribute__((ext_vector_type(8)))  float  v8f;

// ---------- helpers ----------
__device__ __forceinline__ uint32_t pack_bf16x2(float a, float b) {
    uint32_t ua = __float_as_uint(a);
    uint32_t ub = __float_as_uint(b);
    ua = (ua + 0x7FFFu + ((ua >> 16) & 1u)) >> 16;   // RNE truncate to bf16
    ub = (ub + 0x7FFFu + ((ub >> 16) & 1u)) >> 16;
    return (ub << 16) | (ua & 0xFFFFu);
}
__device__ __forceinline__ uint16_t bf16bits(float a) {
    uint32_t u = __float_as_uint(a);
    return (uint16_t)((u + 0x7FFFu + ((u >> 16) & 1u)) >> 16);
}
// order-preserving float<->uint encoding for atomicMax on floats
__device__ __forceinline__ uint32_t enc_ordered(float f) {
    uint32_t b = __float_as_uint(f);
    return (b & 0x80000000u) ? ~b : (b | 0x80000000u);
}
__device__ __forceinline__ float dec_ordered(uint32_t u) {
    if (u == 0u) return 0.f;
    uint32_t b = (u & 0x80000000u) ? (u ^ 0x80000000u) : ~u;
    return __uint_as_float(b);
}
__device__ __forceinline__ float lrelu(float e) { return e > 0.f ? e : NEG_SLOPE * e; }

// B operand: prepacked lane layout [kt][nt][lane][8 dwords]; 32B per lane
__device__ __forceinline__ v16bf load_b_tile(const uint32_t* __restrict__ B, int tile_idx, int lane) {
    union { v16bf v; uint4 q[2]; } u;
    const uint4* p = reinterpret_cast<const uint4*>(B + ((size_t)tile_idx * 32 + lane) * 8);
    u.q[0] = p[0];
    u.q[1] = p[1];
    return u.v;
}
// A fragment for one 16x32 k-tile from a packed-bf16 row (dword pointer to row start).
// Per-lane K set is {kb..kb+7} U {kb+16..kb+23}: two aligned 128-bit loads.
__device__ __forceinline__ v16bf load_a_frag(const uint32_t* __restrict__ row_dw, int kt, int kb2) {
    union { v16bf v; uint4 q[2]; } u;
    const uint4* p = reinterpret_cast<const uint4*>(row_dw + kt * 16 + kb2);
    u.q[0] = p[0];   // K = kb .. kb+7
    u.q[1] = p[2];   // K = kb+16 .. kb+23
    return u.v;
}

// ---------- weight prep: f32 row-major [K][Ncols] -> bf16 WMMA-B lane layout ----------
__global__ void prep_b_kernel(const float* __restrict__ W, uint32_t* __restrict__ Bout,
                              int K, int Ncols, int ktiles, int ntiles) {
    int tid = blockIdx.x * blockDim.x + threadIdx.x;
    int total = ktiles * ntiles * 32;
    if (tid >= total) return;
    int lane = tid & 31;
    int tile = tid >> 5;                 // kt*ntiles + nt
    int nt = tile % ntiles;
    int kt = tile / ntiles;
    int n = nt * 16 + (lane & 15);
    int kbase = kt * 32 + ((lane < 16) ? 0 : 8);
    uint32_t* dst = Bout + (size_t)tid * 8;
#pragma unroll
    for (int d = 0; d < 8; ++d) {
        int k = kbase + ((d < 4) ? 2 * d : 16 + 2 * (d - 4));
        float a = (k     < K && n < Ncols) ? W[(size_t)k       * Ncols + n] : 0.f;
        float b = ((k+1) < K && n < Ncols) ? W[(size_t)(k + 1) * Ncols + n] : 0.f;
        dst[d] = pack_bf16x2(a, b);
    }
}

__global__ void zero_kernel(uint32_t* __restrict__ p, size_t n) {
    size_t i = (size_t)blockIdx.x * blockDim.x + threadIdx.x;
    if (i < n) p[i] = 0u;
}

// ---------- x f32 [N][166] -> packed bf16 [N][96 dwords] (K padded to 192) ----------
__global__ void cvt_x_kernel(const float* __restrict__ x, uint32_t* __restrict__ xb) {
    size_t i = (size_t)blockIdx.x * blockDim.x + threadIdx.x;
    if (i >= (size_t)N_NODES * (KP_IN / 2)) return;
    int row = (int)(i / (KP_IN / 2));
    int k   = (int)(i % (KP_IN / 2)) * 2;
    float a = 0.f, b = 0.f;
    if (k + 1 < F_IN) {
        float2 v = *reinterpret_cast<const float2*>(x + (size_t)row * F_IN + k);
        a = v.x; b = v.y;
    }
    xb[i] = pack_bf16x2(a, b);
}

// ---------- outb = bf16(out_acc + b_gat): [N][64 dwords] ----------
__global__ void cvt_out_kernel(const float* __restrict__ out_acc, const float* __restrict__ b_gat,
                               uint32_t* __restrict__ outb) {
    size_t i = (size_t)blockIdx.x * blockDim.x + threadIdx.x;
    if (i >= (size_t)N_NODES * (HID / 2)) return;
    int dp = (int)(i % (HID / 2));
    float2 v = reinterpret_cast<const float2*>(out_acc)[i];
    float2 g = reinterpret_cast<const float2*>(b_gat)[dp];
    outb[i] = pack_bf16x2(v.x + g.x, v.y + g.y);
}

// ---------- GEMM1: h = relu(xb @ W_in + b_in) -> bf16 [N][128] ----------
__global__ void gemm_in_kernel(const uint32_t* __restrict__ xb, const uint32_t* __restrict__ Bin,
                               const float* __restrict__ b_in, uint16_t* __restrict__ h16) {
    const int lane = threadIdx.x & 31;
    const int nt   = threadIdx.x >> 5;       // wave == n-tile 0..7
    const int tile_m = blockIdx.x * 16;      // one m-tile per block
    const int kb2 = (lane < 16) ? 0 : 4;     // kb/2 in dwords
    const uint32_t* row_dw = xb + (size_t)(tile_m + (lane & 15)) * (KP_IN / 2);
    v8f c = {};
#pragma unroll
    for (int kt = 0; kt < 6; ++kt) {
        v16bf a = load_a_frag(row_dw, kt, kb2);
        v16bf b = load_b_tile(Bin, kt * 8 + nt, lane);
        c = __builtin_amdgcn_wmma_f32_16x16x32_bf16(false, a, false, b, (short)0, c, false, false);
    }
    const int col = nt * 16 + (lane & 15);
    const float bias = b_in[col];
    const int rbase = (lane < 16) ? 0 : 8;
#pragma unroll
    for (int r = 0; r < 8; ++r) {
        float v = c[r] + bias;
        v = v > 0.f ? v : 0.f;
        h16[(size_t)(tile_m + rbase + r) * HID + col] = bf16bits(v);
    }
}

// ---------- GEMM2: hW = h @ W_gat -> f32 [N][128] ----------
__global__ void gemm_hw_kernel(const uint16_t* __restrict__ h16, const uint32_t* __restrict__ Bgat,
                               float* __restrict__ hW) {
    const int lane = threadIdx.x & 31;
    const int nt   = threadIdx.x >> 5;
    const int tile_m = blockIdx.x * 16;
    const int kb2 = (lane < 16) ? 0 : 4;
    const uint32_t* row_dw =
        reinterpret_cast<const uint32_t*>(h16) + (size_t)(tile_m + (lane & 15)) * (HID / 2);
    v8f c = {};
#pragma unroll
    for (int kt = 0; kt < 4; ++kt) {
        v16bf a = load_a_frag(row_dw, kt, kb2);
        v16bf b = load_b_tile(Bgat, kt * 8 + nt, lane);
        c = __builtin_amdgcn_wmma_f32_16x16x32_bf16(false, a, false, b, (short)0, c, false, false);
    }
    const int col = nt * 16 + (lane & 15);
    const int rbase = (lane < 16) ? 0 : 8;
#pragma unroll
    for (int r = 0; r < 8; ++r)
        hW[(size_t)(tile_m + rbase + r) * HID + col] = c[r];
}

// ---------- attention scalars: a_src/a_dst [N][4] ----------
__global__ void att_kernel(const float* __restrict__ hW,
                           const float* __restrict__ att_src, const float* __restrict__ att_dst,
                           float* __restrict__ a_src, float* __restrict__ a_dst) {
    int t = blockIdx.x * blockDim.x + threadIdx.x;
    if (t >= N_NODES * HEADS) return;
    int n = t >> 2, h = t & 3;
    const float4* hp  = reinterpret_cast<const float4*>(hW + (size_t)n * HID + h * 32);
    const float4* asp = reinterpret_cast<const float4*>(att_src + h * 32);
    const float4* adp = reinterpret_cast<const float4*>(att_dst + h * 32);
    float ss = 0.f, dd = 0.f;
#pragma unroll
    for (int i = 0; i < 8; ++i) {
        float4 v = hp[i], a = asp[i], b = adp[i];
        ss += v.x * a.x + v.y * a.y + v.z * a.z + v.w * a.w;
        dd += v.x * b.x + v.y * b.y + v.z * b.z + v.w * b.w;
    }
    a_src[t] = ss;
    a_dst[t] = dd;
}

// ---------- edge pass 1: segment max (ordered-uint atomicMax) ----------
__global__ void edge_max_kernel(const int* __restrict__ ei, int E, int Ep,
                                const float* __restrict__ a_src, const float* __restrict__ a_dst,
                                uint32_t* __restrict__ m_u) {
    int e = blockIdx.x * blockDim.x + threadIdx.x;
    if (e >= Ep) return;
    int sn = (e < E) ? ei[e]     : (e - E);
    int dn = (e < E) ? ei[E + e] : (e - E);
    float4 as = reinterpret_cast<const float4*>(a_src)[sn];
    float4 ad = reinterpret_cast<const float4*>(a_dst)[dn];
    float ev[4] = { lrelu(as.x + ad.x), lrelu(as.y + ad.y), lrelu(as.z + ad.z), lrelu(as.w + ad.w) };
#pragma unroll
    for (int h = 0; h < 4; ++h)
        atomicMax(&m_u[(size_t)dn * 4 + h], enc_ordered(ev[h]));
}

// ---------- edge pass 2: segment sum of exp(e - m) ----------
__global__ void edge_sum_kernel(const int* __restrict__ ei, int E, int Ep,
                                const float* __restrict__ a_src, const float* __restrict__ a_dst,
                                const uint32_t* __restrict__ m_u, float* __restrict__ s_sum) {
    int e = blockIdx.x * blockDim.x + threadIdx.x;
    if (e >= Ep) return;
    int sn = (e < E) ? ei[e]     : (e - E);
    int dn = (e < E) ? ei[E + e] : (e - E);
    float4 as = reinterpret_cast<const float4*>(a_src)[sn];
    float4 ad = reinterpret_cast<const float4*>(a_dst)[dn];
    float ev[4] = { lrelu(as.x + ad.x), lrelu(as.y + ad.y), lrelu(as.z + ad.z), lrelu(as.w + ad.w) };
#pragma unroll
    for (int h = 0; h < 4; ++h) {
        float m = dec_ordered(m_u[(size_t)dn * 4 + h]);
        atomicAdd(&s_sum[(size_t)dn * 4 + h], __expf(ev[h] - m));
    }
}

// ---------- edge pass 3: message scatter, one wave per edge ----------
__global__ void edge_msg_kernel(const int* __restrict__ ei, int E, int Ep,
                                const float* __restrict__ a_src, const float* __restrict__ a_dst,
                                const uint32_t* __restrict__ m_u, const float* __restrict__ s_sum,
                                const float* __restrict__ hW, float* __restrict__ out_acc) {
    int gw = (blockIdx.x * blockDim.x + threadIdx.x) >> 5;   // global wave id == edge id
    int lane = threadIdx.x & 31;
    if (gw >= Ep) return;
    int sn = (gw < E) ? ei[gw]     : (gw - E);
    int dn = (gw < E) ? ei[E + gw] : (gw - E);
    float4 as = reinterpret_cast<const float4*>(a_src)[sn];
    float4 ad = reinterpret_cast<const float4*>(a_dst)[dn];
    float ev[4] = { lrelu(as.x + ad.x), lrelu(as.y + ad.y), lrelu(as.z + ad.z), lrelu(as.w + ad.w) };
    float alpha[4];
#pragma unroll
    for (int h = 0; h < 4; ++h) {
        float m = dec_ordered(m_u[(size_t)dn * 4 + h]);
        float ss = s_sum[(size_t)dn * 4 + h];
        alpha[h] = __expf(ev[h] - m) / (ss + 1e-16f);
    }
#pragma unroll
    for (int h = 0; h < 4; ++h) {
        int col = h * 32 + lane;
        atomicAdd(&out_acc[(size_t)dn * HID + col], hW[(size_t)sn * HID + col] * alpha[h]);
    }
}

// ---------- fused tail: outb @ W1 + b1, relu, @ W2 + b2 -> logits ----------
__global__ void final_kernel(const uint32_t* __restrict__ outb,
                             const uint32_t* __restrict__ B1, const float* __restrict__ b1,
                             const uint32_t* __restrict__ B2, const float* __restrict__ b2,
                             float* __restrict__ out) {
    __shared__ uint16_t lds_h2[8][16 * 64];
    const int lane = threadIdx.x & 31;
    const int wv   = threadIdx.x >> 5;
    const int tile_m = (blockIdx.x * 8 + wv) * 16;
    const bool active = tile_m < N_NODES;
    const int mrow = lane & 15;
    const int kb2  = (lane < 16) ? 0 : 4;
    const int rbase = (lane < 16) ? 0 : 8;

    if (active) {
        v8f zero = {};
        v8f acc[4] = { zero, zero, zero, zero };
        const uint32_t* row_dw = outb + (size_t)(tile_m + mrow) * (HID / 2);
#pragma unroll
        for (int kt = 0; kt < 4; ++kt) {
            v16bf a = load_a_frag(row_dw, kt, kb2);
#pragma unroll
            for (int nt = 0; nt < 4; ++nt) {
                v16bf b = load_b_tile(B1, kt * 4 + nt, lane);
                acc[nt] = __builtin_amdgcn_wmma_f32_16x16x32_bf16(false, a, false, b, (short)0,
                                                                  acc[nt], false, false);
            }
        }
#pragma unroll
        for (int nt = 0; nt < 4; ++nt) {
            int col = nt * 16 + (lane & 15);
            float bias = b1[col];
#pragma unroll
            for (int r = 0; r < 8; ++r) {
                float v = acc[nt][r] + bias;
                v = v > 0.f ? v : 0.f;
                lds_h2[wv][(rbase + r) * 64 + col] = bf16bits(v);
            }
        }
    }
    __syncthreads();
    if (active) {
        v8f cl = {};
#pragma unroll
        for (int kt = 0; kt < 2; ++kt) {
            union { v16bf v; uint32_t u[8]; } a;
#pragma unroll
            for (int d = 0; d < 8; ++d) {
                int k = kt * 32 + (kb2 * 2) + ((d < 4) ? 2 * d : 16 + 2 * (d - 4));
                a.u[d] = *reinterpret_cast<const uint32_t*>(&lds_h2[wv][mrow * 64 + k]);
            }
            v16bf b = load_b_tile(B2, kt, lane);
            cl = __builtin_amdgcn_wmma_f32_16x16x32_bf16(false, a.v, false, b, (short)0, cl, false, false);
        }
        int col = lane & 15;
        if (col < 2) {
            float bias = b2[col];
#pragma unroll
            for (int r = 0; r < 8; ++r)
                out[(size_t)(tile_m + rbase + r) * 2 + col] = cl[r] + bias;
        }
    }
}

// ---------- launch ----------
extern "C" void kernel_launch(void* const* d_in, const int* in_sizes, int n_in,
                              void* d_out, int out_size, void* d_ws, size_t ws_size,
                              hipStream_t stream) {
    const float* x       = (const float*)d_in[0];
    const int*   ei      = (const int*)d_in[1];
    const float* W_in    = (const float*)d_in[2];
    const float* b_in    = (const float*)d_in[3];
    const float* W_gat   = (const float*)d_in[4];
    const float* att_src = (const float*)d_in[5];
    const float* att_dst = (const float*)d_in[6];
    const float* b_gat   = (const float*)d_in[7];
    const float* W1      = (const float*)d_in[8];
    const float* b1      = (const float*)d_in[9];
    const float* W2      = (const float*)d_in[10];
    const float* b2      = (const float*)d_in[11];

    const int E  = in_sizes[1] / 2;
    const int Ep = E + N_NODES;

    char* ws = (char*)d_ws;
    size_t off = 0;
    auto take = [&](size_t bytes) -> char* {
        char* p = ws + off;
        off = (off + bytes + 255) & ~(size_t)255;
        return p;
    };
    // zero-initialized region must be contiguous: m_u, s_sum, out_acc
    uint32_t* m_u    = (uint32_t*)take((size_t)N_NODES * 4 * 4);
    float*    s_sum  = (float*)   take((size_t)N_NODES * 4 * 4);
    float*    out_ac = (float*)   take((size_t)N_NODES * HID * 4);
    uint16_t* h16    = (uint16_t*)take((size_t)N_NODES * HID * 2);  // reused as outb later
    float*    hW     = (float*)   take((size_t)N_NODES * HID * 4);
    float*    a_src  = (float*)   take((size_t)N_NODES * 4 * 4);
    float*    a_dst  = (float*)   take((size_t)N_NODES * 4 * 4);
    uint32_t* xb     = (uint32_t*)take((size_t)N_NODES * (KP_IN / 2) * 4);
    uint32_t* B_in   = (uint32_t*)take((size_t)6 * 8 * 32 * 8 * 4);
    uint32_t* B_gat  = (uint32_t*)take((size_t)4 * 8 * 32 * 8 * 4);
    uint32_t* B_W1   = (uint32_t*)take((size_t)4 * 4 * 32 * 8 * 4);
    uint32_t* B_W2   = (uint32_t*)take((size_t)2 * 1 * 32 * 8 * 4);
    uint32_t* outb   = (uint32_t*)h16;   // h16 is dead once hW exists; same 25.6MB footprint

    prep_b_kernel<<<(6 * 8 * 32 + 255) / 256, 256, 0, stream>>>(W_in,  B_in,  F_IN, HID, 6, 8);
    prep_b_kernel<<<(4 * 8 * 32 + 255) / 256, 256, 0, stream>>>(W_gat, B_gat, HID,  HID, 4, 8);
    prep_b_kernel<<<(4 * 4 * 32 + 255) / 256, 256, 0, stream>>>(W1,    B_W1,  HID,  64,  4, 4);
    prep_b_kernel<<<(2 * 1 * 32 + 255) / 256, 256, 0, stream>>>(W2,    B_W2,  64,   2,   2, 1);

    size_t nz = (size_t)N_NODES * (4 + 4 + HID);
    zero_kernel<<<(unsigned)((nz + 255) / 256), 256, 0, stream>>>(m_u, nz);

    size_t nxb = (size_t)N_NODES * (KP_IN / 2);
    cvt_x_kernel<<<(unsigned)((nxb + 255) / 256), 256, 0, stream>>>(x, xb);

    gemm_in_kernel<<<N_NODES / 16, 256, 0, stream>>>(xb, B_in, b_in, h16);
    gemm_hw_kernel<<<N_NODES / 16, 256, 0, stream>>>(h16, B_gat, hW);
    att_kernel<<<(N_NODES * HEADS + 255) / 256, 256, 0, stream>>>(hW, att_src, att_dst, a_src, a_dst);

    edge_max_kernel<<<(Ep + 255) / 256, 256, 0, stream>>>(ei, E, Ep, a_src, a_dst, m_u);
    edge_sum_kernel<<<(Ep + 255) / 256, 256, 0, stream>>>(ei, E, Ep, a_src, a_dst, m_u, s_sum);
    edge_msg_kernel<<<(Ep + 7) / 8, 256, 0, stream>>>(ei, E, Ep, a_src, a_dst, m_u, s_sum, hW, out_ac);

    size_t nob = (size_t)N_NODES * (HID / 2);
    cvt_out_kernel<<<(unsigned)((nob + 255) / 256), 256, 0, stream>>>(out_ac, b_gat, outb);

    final_kernel<<<(N_NODES / 16 + 7) / 8, 256, 0, stream>>>(outb, B_W1, b1, B_W2, b2,
                                                             (float*)d_out);
}
